// PerlinAttention_49795850830557
// MI455X (gfx1250) — compile-verified, matrix operand order checked by reading, not run on hardware
//
#include <hip/hip_runtime.h>
#include <hip/hip_bf16.h>
#include <stdint.h>

// ---------------- problem constants ----------------
#define N_    2
#define H_    16
#define T_    1024
#define HID_  64
#define DH_   1024          // H_*HID_
#define NBF_  266
#define FP_   288           // NBF padded to multiple of 32
#define L_    128
#define TOPK_ 32
#define NH_   (N_*H_)       // 32

typedef _Float16 h16;
typedef __attribute__((ext_vector_type(16))) _Float16 v16h;
typedef __attribute__((ext_vector_type(8)))  float    v8f;
typedef int b128v __attribute__((vector_size(16)));   // matches builtin prototype

#define RATIO_ 0.0613139406f   // 266^-0.5
#define DN_    0.3535533906f   // 64^-0.25
#define DN2_   0.125f          // DN_^2

// ---------------- CDNA5 async-LDS staging (guarded) ----------------
#if defined(__has_builtin)
#  if __has_builtin(__builtin_amdgcn_global_load_async_to_lds_b128)
#    define ASYNC_OK 1
#  endif
#endif
#ifndef ASYNC_OK
#define ASYNC_OK 0
#endif

__device__ __forceinline__ void async_copy16(const void* g, void* l) {
#if ASYNC_OK
  __builtin_amdgcn_global_load_async_to_lds_b128(
      (__attribute__((address_space(1))) b128v*)(uintptr_t)g,
      (__attribute__((address_space(3))) b128v*)(uintptr_t)l,
      0, 0);
#else
  *(uint4*)l = *(const uint4*)g;   // global_load_b128 + ds_store_b128
#endif
}

__device__ __forceinline__ void wait_async() {
#if ASYNC_OK
#  if __has_builtin(__builtin_amdgcn_s_wait_asynccnt)
  __builtin_amdgcn_s_wait_asynccnt(0);
#  else
  asm volatile("s_wait_asynccnt 0" ::: "memory");
#  endif
#endif
}

// ---------------- block reductions (wave32, blockDim=256 -> 8 waves) ----------------
__device__ __forceinline__ float warpRedSum(float v) {
#pragma unroll
  for (int o = 16; o > 0; o >>= 1) v += __shfl_xor(v, o, 32);
  return v;
}
__device__ __forceinline__ float warpRedMax(float v) {
#pragma unroll
  for (int o = 16; o > 0; o >>= 1) v = fmaxf(v, __shfl_xor(v, o, 32));
  return v;
}
__device__ __forceinline__ float blockSum(float v, float* scr) {
  int lane = threadIdx.x & 31, w = threadIdx.x >> 5;
  v = warpRedSum(v);
  if (lane == 0) scr[w] = v;
  __syncthreads();
  float x = (lane < 8) ? scr[lane] : 0.0f;
  x = warpRedSum(x);
  __syncthreads();
  return x;
}
__device__ __forceinline__ float blockMax(float v, float* scr) {
  int lane = threadIdx.x & 31, w = threadIdx.x >> 5;
  v = warpRedMax(v);
  if (lane == 0) scr[w] = v;
  __syncthreads();
  float x = (lane < 8) ? scr[lane] : -3.0e38f;
  x = warpRedMax(x);
  __syncthreads();
  return x;
}

// ---------------- batched WMMA GEMM ----------------
// C[M,Nn] = alpha * A[M,K] * op(B),  f16 inputs, f32 out.
// B_KN=false: B stored [Nn,K] row-major (NT).  B_KN=true: B stored [K,Nn] (NN),
// transposed into LDS during staging so fragment reads are identical.
template <bool B_KN>
__global__ __launch_bounds__(256) void gemm_wmma(
    const h16* __restrict__ A, long long strA, int lda,
    const h16* __restrict__ B, long long strB, int ldb,
    float* __restrict__ C, long long strC, int ldc,
    int M, int Nn, int K, float alpha) {
  __shared__ alignas(32) h16 tA[64 * 32];
  __shared__ alignas(32) h16 tB[32 * 32];
  const int tid = threadIdx.x;
  const int lane = tid & 31, wave = tid >> 5;
  const int wm = (wave & 3) * 16, wn = (wave >> 2) * 16;
  const long long z = blockIdx.z;
  A += z * strA; B += z * strB; C += z * strC;
  const int bm = blockIdx.y * 64, bn = blockIdx.x * 32;
  v8f acc = {};
  for (int k0 = 0; k0 < K; k0 += 32) {
    {  // stage A tile 64x32 (256 x 16B)
      int r = tid >> 2, c = (tid & 3) * 8;
      int ra = bm + r; if (ra > M - 1) ra = M - 1;
      async_copy16(A + (long long)ra * lda + (k0 + c), &tA[r * 32 + c]);
      if (k0 + 32 < K) __builtin_prefetch(A + (long long)ra * lda + (k0 + 32 + c), 0, 1);
    }
    if (B_KN) {  // B is [K,Nn]; transpose into tB[N][K]
      if (tid < 128) {
        int kk = tid >> 2, c = tid & 3;
        h16 tmp[8];
        *(uint4*)tmp = *(const uint4*)(B + (long long)(k0 + kk) * ldb + (bn + c * 8));
#pragma unroll
        for (int j = 0; j < 8; ++j) tB[(c * 8 + j) * 32 + kk] = tmp[j];
      }
    } else {     // B is [Nn,K]
      if (tid < 128) {
        int r = tid >> 2, c = (tid & 3) * 8;
        int rb2 = bn + r; if (rb2 > Nn - 1) rb2 = Nn - 1;
        async_copy16(B + (long long)rb2 * ldb + (k0 + c), &tB[r * 32 + c]);
      }
    }
    wait_async();
    __syncthreads();
    v16h av = *(const v16h*)&tA[(wm + (lane & 15)) * 32 + (lane >> 4) * 16];
    v16h bv = *(const v16h*)&tB[(wn + (lane & 15)) * 32 + (lane >> 4) * 16];
    acc = __builtin_amdgcn_wmma_f32_16x16x32_f16(false, av, false, bv,
                                                 (short)0, acc, false, false);
    __syncthreads();
  }
  // C layout: VGPR i -> M = i + (lane>=16 ? 8 : 0), N = lane&15
  int cn = bn + wn + (lane & 15);
  int rb = bm + wm + (lane >> 4) * 8;
  if (cn < Nn) {
#pragma unroll
    for (int i = 0; i < 8; ++i) {
      int rm = rb + i;
      if (rm < M) C[(long long)rm * ldc + cn] = alpha * acc[i];
    }
  }
}

// ---------------- conversions ----------------
__global__ void k_cvt16(h16* __restrict__ d, const float* __restrict__ s, long long n) {
  long long i = (long long)blockIdx.x * blockDim.x + threadIdx.x;
  long long st = (long long)gridDim.x * blockDim.x;
  for (; i < n; i += st) d[i] = (h16)s[i];
}
// v * (mask > -1)   over (N,H,T,HID)
__global__ void k_cvt16_vmask(h16* __restrict__ d, const float* __restrict__ s,
                              const float* __restrict__ mask) {
  const long long n = (long long)N_ * H_ * T_ * HID_;
  long long i = (long long)blockIdx.x * blockDim.x + threadIdx.x;
  long long st = (long long)gridDim.x * blockDim.x;
  for (; i < n; i += st) {
    long long t = (i >> 6) % T_;
    long long n0 = i / ((long long)H_ * T_ * HID_);
    float m = (mask[n0 * T_ + t] > -1.0f) ? 1.0f : 0.0f;
    d[i] = (h16)(s[i] * m);
  }
}
// proj [266,64] -> padded f16 [288,64]
__global__ void k_cvt_proj(h16* __restrict__ d, const float* __restrict__ s) {
  int i = blockIdx.x * 256 + threadIdx.x;
  if (i < FP_ * HID_) {
    int f = i >> 6;
    d[i] = (f < NBF_) ? (h16)s[i] : (h16)0.0f;
  }
}

// ---------------- performer feature kernels ----------------
__global__ __launch_bounds__(256) void k_feat_q(h16* __restrict__ qp,
                                                const float* __restrict__ dash,
                                                const float* __restrict__ xa) {
  __shared__ float scr[8];
  long long row = blockIdx.x;  // nh*T rows
  const float* dr = dash + row * FP_;
  int tid = threadIdx.x;
  float lm = -1e30f;
  for (int f = tid; f < NBF_; f += 256) lm = fmaxf(lm, dr[f]);
  float m = blockMax(lm, scr);
  const float* xr = xa + row * HID_;
  float ss = 0.0f;
  if (tid < HID_) { float v = xr[tid]; ss = v * v; }
  float diag = 0.5f * DN2_ * blockSum(ss, scr);
  h16* qr = qp + row * FP_;
  for (int f = tid; f < FP_; f += 256)
    qr[f] = (f < NBF_) ? (h16)(RATIO_ * (expf(dr[f] - diag - m) + 1e-4f)) : (h16)0.0f;
}

__global__ __launch_bounds__(256) void k_rowmax(float* __restrict__ rmax,
                                                const float* __restrict__ dash) {
  __shared__ float scr[8];
  long long r = blockIdx.x;
  const float* dr = dash + r * FP_;
  float lm = -1e30f;
  for (int f = threadIdx.x; f < NBF_; f += 256) lm = fmaxf(lm, dr[f]);
  float m = blockMax(lm, scr);
  if (threadIdx.x == 0) rmax[r] = m;
}
__global__ __launch_bounds__(256) void k_colmax(float* __restrict__ kmax,
                                                const float* __restrict__ rmax) {
  __shared__ float scr[8];
  int nh = blockIdx.x;
  float lm = -1e30f;
  for (int t = threadIdx.x; t < T_; t += 256) lm = fmaxf(lm, rmax[nh * T_ + t]);
  float m = blockMax(lm, scr);
  if (threadIdx.x == 0) kmax[nh] = m;
}
__global__ __launch_bounds__(256) void k_feat_k(h16* __restrict__ kp,
                                                const float* __restrict__ dash,
                                                const float* __restrict__ xa,
                                                const float* __restrict__ kmax) {
  __shared__ float scr[8];
  long long row = blockIdx.x;
  float m = kmax[row >> 10];
  const float* dr = dash + row * FP_;
  int tid = threadIdx.x;
  const float* xr = xa + row * HID_;
  float ss = 0.0f;
  if (tid < HID_) { float v = xr[tid]; ss = v * v; }
  float diag = 0.5f * DN2_ * blockSum(ss, scr);
  h16* kr = kp + row * FP_;
  for (int f = tid; f < FP_; f += 256)
    kr[f] = (f < NBF_) ? (h16)(RATIO_ * (expf(dr[f] - diag - m) + 1e-4f)) : (h16)0.0f;
}
// ksum[nh][f] = sum_t kp[nh][t][f]
__global__ void k_ksum(float* __restrict__ ksum, const h16* __restrict__ kp) {
  int nh = blockIdx.x;
  int f = blockIdx.y * 256 + threadIdx.x;
  if (f >= FP_) return;
  const h16* base = kp + (long long)nh * T_ * FP_ + f;
  float s = 0.0f;
  for (int t = 0; t < T_; ++t) s += (float)base[(long long)t * FP_];
  ksum[nh * FP_ + f] = s;
}
// kpT[nh][f][t] = kp[nh][t][f]
__global__ void k_transpose_kp(h16* __restrict__ kt, const h16* __restrict__ kp) {
  long long nh = blockIdx.z;
  int t = blockIdx.x * 256 + threadIdx.x;
  int f = blockIdx.y;
  kt[(nh * FP_ + f) * (long long)T_ + t] = kp[(nh * T_ + t) * (long long)FP_ + f];
}
// perf *= 1 / (qp . ksum)
__global__ __launch_bounds__(256) void k_dinv(float* __restrict__ perf,
                                              const h16* __restrict__ qp,
                                              const float* __restrict__ ksum) {
  __shared__ float scr[8];
  long long row = blockIdx.x;
  int nh = (int)(row >> 10);
  const h16* qr = qp + row * FP_;
  const float* ks = ksum + nh * FP_;
  float s = 0.0f;
  for (int f = threadIdx.x; f < NBF_; f += 256) s += (float)qr[f] * ks[f];
  float inv = 1.0f / blockSum(s, scr);
  float* pr = perf + row * HID_;
  if (threadIdx.x < HID_) pr[threadIdx.x] *= inv;
}

// ---------------- MLP path ----------------
// X[n*T+t][h*128 + j] = j<64 ? perf[n,h,t,j] : v[n,h,t,j-64]*vmask
__global__ __launch_bounds__(256) void k_buildX(h16* __restrict__ X,
                                                const float* __restrict__ perf,
                                                const float* __restrict__ v,
                                                const float* __restrict__ mask) {
  int row = blockIdx.x;  // N*T
  int n = row >> 10, t = row & 1023;
  h16* xr = X + (long long)row * (2 * DH_);
  float vm = (mask[n * T_ + t] > -1.0f) ? 1.0f : 0.0f;
  for (int c = threadIdx.x; c < 2 * DH_; c += 256) {
    int h = c >> 7, j = c & 127;
    long long base = (((long long)(n * H_ + h)) * T_ + t) * HID_;
    float val = (j < HID_) ? perf[base + j] : v[base + (j - 64)] * vm;
    xr[c] = (h16)val;
  }
}
__global__ __launch_bounds__(256) void k_ln_gelu(float* __restrict__ t32, h16* __restrict__ t16,
                                                 const float* __restrict__ enc,
                                                 const float* __restrict__ eb,
                                                 const float* __restrict__ g,
                                                 const float* __restrict__ b) {
  __shared__ float scr[8];
  int row = blockIdx.x;  // N*T
  const float* er = enc + (long long)row * DH_;
  float x[4];
  float s = 0.0f;
#pragma unroll
  for (int i = 0; i < 4; ++i) {
    int c = threadIdx.x + i * 256;
    x[i] = er[c] + eb[c];
    s += x[i];
  }
  float mu = blockSum(s, scr) * (1.0f / DH_);
  float vs = 0.0f;
#pragma unroll
  for (int i = 0; i < 4; ++i) { float d = x[i] - mu; vs += d * d; }
  float inv = rsqrtf(blockSum(vs, scr) * (1.0f / DH_) + 1e-5f);
#pragma unroll
  for (int i = 0; i < 4; ++i) {
    int c = threadIdx.x + i * 256;
    float y = (x[i] - mu) * inv * g[c] + b[c];
    float ge = 0.5f * y * (1.0f + erff(y * 0.70710678f));
    t32[(long long)row * DH_ + c] = ge;
    t16[(long long)row * DH_ + c] = (h16)ge;
  }
}
__global__ __launch_bounds__(256) void k_scaler(float* __restrict__ sc,
                                                const float* __restrict__ t32,
                                                const float* __restrict__ w,
                                                const float* __restrict__ b) {
  __shared__ float scr[8];
  int row = blockIdx.x;  // N*T
  const float* tr = t32 + (long long)row * DH_;
  for (int h = 0; h < H_; ++h) {
    const float* wr = w + h * DH_;
    float s = 0.0f;
    for (int c = threadIdx.x; c < DH_; c += 256) s += tr[c] * wr[c];
    float d = blockSum(s, scr);
    if (threadIdx.x == 0) sc[(long long)row * H_ + h] = 1.0f / (1.0f + expf(-(d + b[h])));
  }
}

// ---------------- est: resize + mask + softmax + KL/MSE partials ----------------
__global__ __launch_bounds__(256) void k_est(float* __restrict__ est_probs,
                                             float* __restrict__ klp, float* __restrict__ msep,
                                             const float* __restrict__ dec,
                                             const float* __restrict__ drb,
                                             const float* __restrict__ mask,
                                             const float* __restrict__ truth) {
  __shared__ float scr[8];
  __shared__ float eL[L_];
  long long r = blockIdx.x;  // N*H*T
  int n = (int)(r >> 14), h = (int)((r >> 10) & 15), t = (int)(r & 1023);
  int tid = threadIdx.x;
  if (tid < L_)
    eL[tid] = dec[((long long)(n * T_ + t)) * (H_ * L_) + h * L_ + tid] + drb[h * L_ + tid];
  __syncthreads();
  const float* mrow = mask + n * T_;
  const float* trow = truth + r * (long long)T_;
  float e[4], tr[4];
  float lmax = -1e30f;
#pragma unroll
  for (int i = 0; i < 4; ++i) {
    int s = tid + i * 256;
    float x = (s + 0.5f) * 0.125f - 0.5f;
    float fl = floorf(x);
    int l0 = (int)fl;
    float w = x - fl;
    int la = l0 < 0 ? 0 : l0;
    int lb = (l0 + 1 > L_ - 1) ? (L_ - 1) : (l0 + 1);
    e[i] = eL[la] * (1.0f - w) + eL[lb] * w + mrow[s];
    tr[i] = trow[s];
    lmax = fmaxf(lmax, e[i]);
  }
  float m = blockMax(lmax, scr);
  float ex[4];
  float es = 0.0f;
#pragma unroll
  for (int i = 0; i < 4; ++i) { ex[i] = expf(e[i] - m); es += ex[i]; }
  float sum = blockSum(es, scr);
  float inv = 1.0f / sum, lsum = logf(sum);
  float tm = -1e30f;
#pragma unroll
  for (int i = 0; i < 4; ++i) tm = fmaxf(tm, tr[i]);
  tm = blockMax(tm, scr);
  float tx[4];
  float ts = 0.0f;
#pragma unroll
  for (int i = 0; i < 4; ++i) { tx[i] = expf(tr[i] - tm); ts += tx[i]; }
  float tinv = 1.0f / blockSum(ts, scr);
  float* op = est_probs + r * (long long)T_;
  float klacc = 0.0f, mseacc = 0.0f;
#pragma unroll
  for (int i = 0; i < 4; ++i) {
    int s = tid + i * 256;
    op[s] = ex[i] * inv;
    float logq = (e[i] - m) - lsum;
    float p = tx[i] * tinv;
    float plogp = (p > 0.0f) ? p * logf(fmaxf(p, 1e-20f)) : 0.0f;
    if (mrow[s] > -1.0f) klacc += plogp - p * logq;
    float d = e[i] - tr[i];
    mseacc += d * d;
  }
  float klb = blockSum(klacc, scr);
  float mb = blockSum(mseacc, scr);
  if (tid == 0) { klp[r] = klb; msep[r] = mb; }
}

// ---------------- dense softmax + MSE partials ----------------
__global__ __launch_bounds__(256) void k_dense_sm(float* __restrict__ probs_dense,
                                                  float* __restrict__ msed,
                                                  const float* __restrict__ draw,
                                                  const float* __restrict__ mask,
                                                  const float* __restrict__ truth) {
  __shared__ float scr[8];
  long long r = blockIdx.x;
  int n = (int)(r >> 14);
  int tid = threadIdx.x;
  const float* mrow = mask + n * T_;
  const float* dr = draw + r * (long long)T_;
  const float* trow = truth + r * (long long)T_;
  float d[4];
  float lmax = -1e30f;
#pragma unroll
  for (int i = 0; i < 4; ++i) {
    int s = tid + i * 256;
    d[i] = dr[s] + mrow[s];
    lmax = fmaxf(lmax, d[i]);
  }
  float m = blockMax(lmax, scr);
  float ex[4];
  float es = 0.0f, mse = 0.0f;
#pragma unroll
  for (int i = 0; i < 4; ++i) {
    ex[i] = expf(d[i] - m);
    es += ex[i];
    int s = tid + i * 256;
    float df = d[i] - trow[s];
    mse += df * df;
  }
  float inv = 1.0f / blockSum(es, scr);
  float* op = probs_dense + r * (long long)T_;
#pragma unroll
  for (int i = 0; i < 4; ++i) { int s = tid + i * 256; op[s] = ex[i] * inv; }
  float mb = blockSum(mse, scr);
  if (tid == 0) msed[r] = mb;
}

// ---------------- per-row top-32 + partial softmax * scale ----------------
__global__ __launch_bounds__(256) void k_partial(float* __restrict__ pp, h16* __restrict__ pp16,
                                                 const float* __restrict__ est_probs,
                                                 const float* __restrict__ draw,
                                                 const float* __restrict__ mask,
                                                 const float* __restrict__ sc) {
  __shared__ float scr[8];
  __shared__ float ep[T_];
  __shared__ int sel;
  long long r = blockIdx.x;
  int n = (int)(r >> 14), h = (int)((r >> 10) & 15), t = (int)(r & 1023);
  int tid = threadIdx.x;
  const float* er = est_probs + r * (long long)T_;
  float eo[4];
#pragma unroll
  for (int i = 0; i < 4; ++i) {
    int s = tid + i * 256;
    eo[i] = er[s];
    ep[s] = eo[i];
  }
  __syncthreads();
  float thr = 1e30f;
  for (int it = 0; it < TOPK_; ++it) {
    float lm = -1e30f;
#pragma unroll
    for (int i = 0; i < 4; ++i) lm = fmaxf(lm, ep[tid + i * 256]);
    float m = blockMax(lm, scr);
    if (tid == 0) sel = 0x7fffffff;
    __syncthreads();
#pragma unroll
    for (int i = 0; i < 4; ++i) {
      int s = tid + i * 256;
      if (ep[s] == m) atomicMin(&sel, s);
    }
    __syncthreads();
    if (tid == 0) ep[sel] = -1e30f;
    thr = m;
    __syncthreads();
  }
  const float* mrow = mask + n * T_;
  const float* dr = draw + r * (long long)T_;
  float pv[4];
  float lmax = -1e30f;
#pragma unroll
  for (int i = 0; i < 4; ++i) {
    int s = tid + i * 256;
    float mv = mrow[s];
    float val = (eo[i] >= thr) ? (dr[s] + mv) : -10000.0f;
    pv[i] = val + mv;
    lmax = fmaxf(lmax, pv[i]);
  }
  float m = blockMax(lmax, scr);
  float ex[4];
  float es = 0.0f;
#pragma unroll
  for (int i = 0; i < 4; ++i) { ex[i] = expf(pv[i] - m); es += ex[i]; }
  float inv = 1.0f / blockSum(es, scr);
  float scale = sc[((long long)(n * T_ + t)) * H_ + h];
  float* orow = pp + r * (long long)T_;
  h16* orow16 = pp16 + r * (long long)T_;
#pragma unroll
  for (int i = 0; i < 4; ++i) {
    int s = tid + i * 256;
    float v = ex[i] * inv * scale;
    orow[s] = v;
    orow16[s] = (h16)v;
  }
}

// ---------------- final loss reduction ----------------
__global__ __launch_bounds__(256) void k_loss(float* __restrict__ out,
                                              const float* __restrict__ klp,
                                              const float* __restrict__ msep,
                                              const float* __restrict__ msed,
                                              const float* __restrict__ mask) {
  __shared__ float scr[8];
  int tid = threadIdx.x;
  const long long R = (long long)NH_ * T_;
  float kl = 0.0f, me = 0.0f, md = 0.0f;
  for (long long i = tid; i < R; i += 256) { kl += klp[i]; me += msep[i]; md += msed[i]; }
  kl = blockSum(kl, scr);
  me = blockSum(me, scr);
  md = blockSum(md, scr);
  float vc = 0.0f;
  for (int i = tid; i < N_ * T_; i += 256) vc += (mask[i] > -1.0f) ? 1.0f : 0.0f;
  vc = blockSum(vc, scr);
  if (tid == 0) {
    double valid = (double)vc * H_ * T_;
    if (valid < 1.0) valid = 1.0;
    double numel = (double)N_ * H_ * T_ * T_;
    out[0] = (float)((double)kl / valid * 0.1 + (double)me / numel + (double)md / numel);
  }
}

// ---------------- host orchestration ----------------
extern "C" void kernel_launch(void* const* d_in, const int* in_sizes, int n_in,
                              void* d_out, int out_size, void* d_ws, size_t ws_size,
                              hipStream_t stream) {
  (void)in_sizes; (void)n_in; (void)out_size; (void)ws_size;
  const float* v     = (const float*)d_in[2];
  const float* qa    = (const float*)d_in[3];
  const float* ka    = (const float*)d_in[4];
  const float* va    = (const float*)d_in[5];
  const float* qsc   = (const float*)d_in[6];
  const float* kscn  = (const float*)d_in[7];
  const float* mask  = (const float*)d_in[8];
  const float* truth = (const float*)d_in[9];
  const float* proj  = (const float*)d_in[11];
  const float* encw  = (const float*)d_in[12];
  const float* encb  = (const float*)d_in[13];
  const float* lng   = (const float*)d_in[14];
  const float* lnb   = (const float*)d_in[15];
  const float* decw  = (const float*)d_in[16];
  const float* decb  = (const float*)d_in[17];
  const float* scw   = (const float*)d_in[18];
  const float* scb   = (const float*)d_in[19];

  const long long EHT = (long long)NH_ * T_;       // 32768 rows
  const long long QKV = EHT * HID_;                // 2,097,152

  char* w = (char*)d_ws;
  auto alloc = [&](size_t bytes) -> char* {
    char* p = w;
    w += (bytes + 255) & ~(size_t)255;
    return p;
  };
  h16* qs16   = (h16*)alloc(QKV * 2);
  h16* ks16   = (h16*)alloc(QKV * 2);
  h16* qa16   = (h16*)alloc(QKV * 2);
  h16* ka16   = (h16*)alloc(QKV * 2);
  h16* va16   = (h16*)alloc(QKV * 2);
  h16* v16    = (h16*)alloc(QKV * 2);
  h16* proj16 = (h16*)alloc((size_t)FP_ * HID_ * 2);
  h16* encw16 = (h16*)alloc((size_t)DH_ * 2 * DH_ * 2);
  h16* decw16 = (h16*)alloc((size_t)H_ * L_ * DH_ * 2);
  float* draw = (float*)alloc((size_t)EHT * T_ * 4);
  float* dashq = (float*)alloc((size_t)EHT * FP_ * 4);
  float* dashk = (float*)alloc((size_t)EHT * FP_ * 4);
  h16* qp16  = (h16*)alloc((size_t)EHT * FP_ * 2);
  h16* kp16  = (h16*)alloc((size_t)EHT * FP_ * 2);
  h16* kpT16 = (h16*)alloc((size_t)EHT * FP_ * 2);
  float* rmax  = (float*)alloc((size_t)EHT * 4);
  float* kmaxb = (float*)alloc((size_t)NH_ * 4);
  float* ksum  = (float*)alloc((size_t)NH_ * FP_ * 4);
  float* kv32  = (float*)alloc((size_t)NH_ * FP_ * HID_ * 4);
  h16* kv16    = (h16*)alloc((size_t)NH_ * FP_ * HID_ * 2);
  float* perf  = (float*)alloc((size_t)QKV * 4);
  h16* X16     = (h16*)alloc((size_t)N_ * T_ * 2 * DH_ * 2);
  float* encout = (float*)alloc((size_t)N_ * T_ * DH_ * 4);
  float* t32    = (float*)alloc((size_t)N_ * T_ * DH_ * 4);
  h16* t16      = (h16*)alloc((size_t)N_ * T_ * DH_ * 2);
  float* decout = (float*)alloc((size_t)N_ * T_ * H_ * L_ * 4);
  float* sc     = (float*)alloc((size_t)N_ * T_ * H_ * 4);
  float* klp  = (float*)alloc((size_t)EHT * 4);
  float* msep = (float*)alloc((size_t)EHT * 4);
  float* msed = (float*)alloc((size_t)EHT * 4);
  h16* pp16   = (h16*)alloc((size_t)EHT * T_ * 2);

  float* out   = (float*)d_out;
  float* ctx   = out + 1;                    // (N,H,T,HID)
  float* pprob = ctx + QKV;                  // (N,H,T,T)
  float* eprob = pprob + EHT * T_;           // (N,H,T,T)
  float* dprob = eprob + EHT * T_;           // (N,H,T,T)

  auto cvt = [&](h16* d, const float* s, long long n) {
    int blocks = (int)((n + 1023) / 1024);
    k_cvt16<<<blocks, 256, 0, stream>>>(d, s, n);
  };
  auto gemm_nt = [&](const h16* A, long long sa, int lda, const h16* B, long long sb, int ldb,
                     float* C, long long scst, int ldc, int M, int Nn, int K, float alpha, int bt) {
    dim3 g((Nn + 31) / 32, (M + 63) / 64, bt);
    gemm_wmma<false><<<g, 256, 0, stream>>>(A, sa, lda, B, sb, ldb, C, scst, ldc, M, Nn, K, alpha);
  };
  auto gemm_nn = [&](const h16* A, long long sa, int lda, const h16* B, long long sb, int ldb,
                     float* C, long long scst, int ldc, int M, int Nn, int K, float alpha, int bt) {
    dim3 g((Nn + 31) / 32, (M + 63) / 64, bt);
    gemm_wmma<true><<<g, 256, 0, stream>>>(A, sa, lda, B, sb, ldb, C, scst, ldc, M, Nn, K, alpha);
  };

  // 1) precision conversions
  cvt(qs16, qsc, QKV);
  cvt(ks16, kscn, QKV);
  cvt(qa16, qa, QKV);
  cvt(ka16, ka, QKV);
  k_cvt16_vmask<<<2048, 256, 0, stream>>>(va16, va, mask);
  k_cvt16_vmask<<<2048, 256, 0, stream>>>(v16, v, mask);
  k_cvt_proj<<<(FP_ * HID_ + 255) / 256, 256, 0, stream>>>(proj16, proj);
  cvt(encw16, encw, (long long)DH_ * 2 * DH_);
  cvt(decw16, decw, (long long)H_ * L_ * DH_);

  // 2) dense scores: qs @ ks^T / sqrt(64)   (batched over nh)
  gemm_nt(qs16, (long long)T_ * HID_, HID_, ks16, (long long)T_ * HID_, HID_,
          draw, (long long)T_ * T_, T_, T_, T_, HID_, 0.125f, NH_);

  // 3) performer feature projections: dash = (x @ proj^T) * dn
  gemm_nt(qa16, (long long)T_ * HID_, HID_, proj16, 0, HID_,
          dashq, (long long)T_ * FP_, FP_, T_, FP_, HID_, DN_, NH_);
  gemm_nt(ka16, (long long)T_ * HID_, HID_, proj16, 0, HID_,
          dashk, (long long)T_ * FP_, FP_, T_, FP_, HID_, DN_, NH_);

  // 4) features
  k_feat_q<<<(int)EHT, 256, 0, stream>>>(qp16, dashq, qa);
  k_rowmax<<<(int)EHT, 256, 0, stream>>>(rmax, dashk);
  k_colmax<<<NH_, 256, 0, stream>>>(kmaxb, rmax);
  k_feat_k<<<(int)EHT, 256, 0, stream>>>(kp16, dashk, ka, kmaxb);
  {
    dim3 g(NH_, (FP_ + 255) / 256, 1);
    k_ksum<<<g, 256, 0, stream>>>(ksum, kp16);
  }
  {
    dim3 g(T_ / 256, FP_, NH_);
    k_transpose_kp<<<g, 256, 0, stream>>>(kpT16, kp16);
  }

  // 5) kv = kp^T @ va   (M=288, N=64, K=1024)
  gemm_nn(kpT16, (long long)FP_ * T_, T_, va16, (long long)T_ * HID_, HID_,
          kv32, (long long)FP_ * HID_, HID_, FP_, HID_, T_, 1.0f, NH_);
  cvt(kv16, kv32, (long long)NH_ * FP_ * HID_);

  // 6) perf = (qp @ kv) * d_inv
  gemm_nn(qp16, (long long)T_ * FP_, FP_, kv16, (long long)FP_ * HID_, HID_,
          perf, (long long)T_ * HID_, HID_, T_, HID_, FP_, 1.0f, NH_);
  k_dinv<<<(int)EHT, 256, 0, stream>>>(perf, qp16, ksum);

  // 7) MLP: X -> encoder -> LN+GELU -> decoder (+scaler)
  k_buildX<<<N_ * T_, 256, 0, stream>>>(X16, perf, v, mask);
  gemm_nt(X16, 0, 2 * DH_, encw16, 0, 2 * DH_, encout, 0, DH_,
          N_ * T_, DH_, 2 * DH_, 1.0f, 1);
  k_ln_gelu<<<N_ * T_, 256, 0, stream>>>(t32, t16, encout, encb, lng, lnb);
  gemm_nt(t16, 0, DH_, decw16, 0, DH_, decout, 0, H_ * L_,
          N_ * T_, H_ * L_, DH_, 1.0f, 1);
  k_scaler<<<N_ * T_, 256, 0, stream>>>(sc, t32, scw, scb);

  // 8) est resize + softmax + loss partials; dense softmax + loss partials
  k_est<<<(int)EHT, 256, 0, stream>>>(eprob, klp, msep, decout, decb, mask, truth);
  k_dense_sm<<<(int)EHT, 256, 0, stream>>>(dprob, msed, draw, mask, truth);

  // 9) top-k -> partial probs (f32 to d_out, f16 for GEMM)
  k_partial<<<(int)EHT, 256, 0, stream>>>(pprob, pp16, eprob, draw, mask, sc);

  // 10) partial context: partial_probs @ v
  gemm_nn(pp16, (long long)T_ * T_, T_, v16, (long long)T_ * HID_, HID_,
          ctx, (long long)T_ * HID_, HID_, T_, HID_, T_, 1.0f, NH_);

  // 11) final loss
  k_loss<<<1, 256, 0, stream>>>(out, klp, msep, msed, mask);
}
